// ExpertsFeedForward_764504178795
// MI455X (gfx1250) — matrix operand order repaired and checked by reading, not compiled, not run on hardware
//
#include <hip/hip_runtime.h>
#include <hip/hip_bf16.h>

// ---------------- constants (match reference) ----------------
#define D_MODEL 1024
#define HIDDEN  4096
#define NEXP    8
#define NTOK    16384           // 4 * 4096 tokens
#define CAP     512             // expert capacity
#define PA      40              // LDS A stride (bf16 elems): 80B rows, 16B aligned
#define PBK     40              // LDS B stride (bf16 elems, K-contiguous): 80B rows
#define ABUF    (64 * PA)       // elements per A buffer
#define BBUF    (128 * PBK)     // elements per B buffer

typedef __bf16 v16bf __attribute__((ext_vector_type(16)));
typedef __bf16 v8bf  __attribute__((ext_vector_type(8)));
typedef __bf16 v4bf  __attribute__((ext_vector_type(4)));
typedef float  v8f   __attribute__((ext_vector_type(8)));

__device__ __forceinline__ float gelu_tanh(float x) {
    float x3 = x * x * x;
    return 0.5f * x * (1.0f + tanhf(0.79788456080286535588f * (x + 0.044715f * x3)));
}

// LDS byte offset of a shared-memory pointer: flat LDS-aperture addresses
// truncate to addr[31:0] (ISA 10.2), so the low 32 bits are the DS address.
__device__ __forceinline__ unsigned lds_off(const void* p) {
    return (unsigned)(unsigned long long)(uintptr_t)p;
}

// CDNA5 async direct-to-LDS copy, 16B per lane, tracked by ASYNCcnt.
__device__ __forceinline__ void async_copy_b128(unsigned lds_byte, const void* gaddr) {
    asm volatile("global_load_async_to_lds_b128 %0, %1, off"
                 :
                 : "v"(lds_byte), "v"(gaddr)
                 : "memory");
}

__device__ __forceinline__ void wait_asynccnt0() {
    asm volatile("s_wait_asynccnt 0x0" ::: "memory");
}

// ---------------- fp32 -> bf16 streaming convert (no transpose; for x) --------
__global__ void __launch_bounds__(256) cvt_bf16_kernel(const float* __restrict__ src,
                                                       __bf16* __restrict__ dst, long n) {
    long i = ((long)blockIdx.x * blockDim.x + threadIdx.x) * 4;
    if (i >= n) return;
    float4 f = *(const float4*)(src + i);
    v4bf o;
    o[0] = (__bf16)f.x; o[1] = (__bf16)f.y; o[2] = (__bf16)f.z; o[3] = (__bf16)f.w;
    *(v4bf*)(dst + i) = o;
}

// ---------------- fp32 [K,N] -> bf16 [N,K] batched transpose-convert ----------
// grid: (K/32, N/32, batch); block 256 = 32 wide x 8 tall
__global__ void __launch_bounds__(256) cvt_transpose_kernel(const float* __restrict__ src,
                                                            __bf16* __restrict__ dst,
                                                            int K, int N) {
    __shared__ float tile[32][33];
    const size_t bOff = (size_t)blockIdx.z * (size_t)K * N;
    const float* s = src + bOff;
    __bf16* d = dst + bOff;
    int kb = blockIdx.x * 32, nb = blockIdx.y * 32;
    int tx = threadIdx.x & 31, ty = threadIdx.x >> 5;
#pragma unroll
    for (int i = 0; i < 4; ++i) {
        int k = ty + i * 8;
        tile[k][tx] = s[(size_t)(kb + k) * N + nb + tx];
    }
    __syncthreads();
#pragma unroll
    for (int i = 0; i < 4; ++i) {
        int n = ty + i * 8;
        d[(size_t)(nb + n) * K + kb + tx] = (__bf16)tile[tx][n];
    }
}

// ---------------- init scratch (counters / index buffers) ----------------
__global__ void init_kernel(int* cnt, int* tok_idx, float* scores) {
    int i = blockIdx.x * blockDim.x + threadIdx.x;
    if (i < 16) cnt[i] = 0;
    if (i < NEXP * CAP) { tok_idx[i] = 0; scores[i] = 0.0f; }
}

// ---------------- router: logits -> softmax, probs stored [E][T] ----------------
__global__ void __launch_bounds__(256) router_kernel(const float* __restrict__ x,
                                                     const float* __restrict__ gw,
                                                     const float* __restrict__ gb,
                                                     const float* __restrict__ temp,
                                                     float* __restrict__ probs) {
    int wave = threadIdx.x >> 5;
    int lane = threadIdx.x & 31;
    int t = blockIdx.x * 8 + wave;
    const float* xr = x + (size_t)t * D_MODEL;
    float acc[NEXP];
#pragma unroll
    for (int e = 0; e < NEXP; ++e) acc[e] = 0.0f;
    for (int i = lane; i < D_MODEL; i += 32) {
        float xv = xr[i];
        const float* g = gw + (size_t)i * NEXP;
#pragma unroll
        for (int e = 0; e < NEXP; ++e) acc[e] += xv * g[e];
    }
#pragma unroll
    for (int off = 16; off >= 1; off >>= 1) {
#pragma unroll
        for (int e = 0; e < NEXP; ++e) acc[e] += __shfl_xor(acc[e], off, 32);
    }
    if (lane == 0) {
        float st = fmaxf(temp[0], 0.1f);
        float lg[NEXP];
        float mx = -1e30f;
#pragma unroll
        for (int e = 0; e < NEXP; ++e) { lg[e] = (acc[e] + gb[e]) / st; mx = fmaxf(mx, lg[e]); }
        float s = 0.0f;
#pragma unroll
        for (int e = 0; e < NEXP; ++e) { lg[e] = __expf(lg[e] - mx); s += lg[e]; }
        float inv = 1.0f / s;
#pragma unroll
        for (int e = 0; e < NEXP; ++e) probs[(size_t)e * NTOK + t] = lg[e] * inv;
    }
}

// ---------------- per-expert top-CAP: radix select + compact ----------------
__global__ void __launch_bounds__(256) topk_kernel(const float* __restrict__ probs,
                                                   int* __restrict__ tok_idx,
                                                   float* __restrict__ scores,
                                                   int* __restrict__ cnt) {
    int e = blockIdx.x;
    const float* p = probs + (size_t)e * NTOK;
    __shared__ unsigned int hist[256];
    __shared__ unsigned int prefix_s;
    __shared__ int remain_s;
    unsigned prefix = 0u;
    int remain = CAP;
    for (int shift = 24; shift >= 0; shift -= 8) {
        for (int i = threadIdx.x; i < 256; i += blockDim.x) hist[i] = 0u;
        __syncthreads();
        unsigned mask_hi = (shift == 24) ? 0u : (0xFFFFFFFFu << (shift + 8));
        for (int t = threadIdx.x; t < NTOK; t += blockDim.x) {
            unsigned key = __float_as_uint(p[t]);
            if ((key & mask_hi) == prefix) atomicAdd(&hist[(key >> shift) & 255u], 1u);
        }
        __syncthreads();
        if (threadIdx.x == 0) {
            int acc = 0;
            int d = 255;
            for (; d > 0; --d) {
                int h = (int)hist[d];
                if (acc + h >= remain) break;
                acc += h;
            }
            prefix_s = prefix | ((unsigned)d << shift);
            remain_s = remain - acc;
        }
        __syncthreads();
        prefix = prefix_s;
        remain = remain_s;
        __syncthreads();
    }
    unsigned kth = prefix;
    for (int t = threadIdx.x; t < NTOK; t += blockDim.x) {
        unsigned key = __float_as_uint(p[t]);
        bool take = false;
        if (key > kth) {
            take = true;
        } else if (key == kth) {
            int old = atomicAdd(&cnt[2 * e + 1], 1);
            if (old < remain) take = true;
        }
        if (take) {
            int pos = atomicAdd(&cnt[2 * e], 1);
            if (pos < CAP) {
                tok_idx[e * CAP + pos] = t;
                scores[e * CAP + pos] = p[t];
            }
        }
    }
}

// ---------------- gather routed tokens (bf16) ----------------
__global__ void __launch_bounds__(256) gather_kernel(const __bf16* __restrict__ xb,
                                                     const int* __restrict__ tok_idx,
                                                     __bf16* __restrict__ tokb) {
    int ec = blockIdx.x;                    // e*CAP + c
    int t = tok_idx[ec] & (NTOK - 1);
    const __bf16* src = xb + (size_t)t * D_MODEL;
    __bf16* dst = tokb + (size_t)ec * D_MODEL;
    int i = threadIdx.x * 4;
    *(uint2*)(dst + i) = *(const uint2*)(src + i);
}

// =====================================================================
// WMMA 64x128-tile core (bf16 in, fp32 acc), double-buffered LDS with
// CDNA5 async direct-to-LDS copies (global_load_async_to_lds_b128).
// 256 threads = 8 waves. 4x8 grid of 16x16 subtiles; wave w owns row
// (w & 3) and 4 subtile columns starting at ((w>>2)*4).
// A staged row-major [64][PA]; B staged from PRE-TRANSPOSED weights
// [128 n][PBK k]; lane fragments are contiguous 16B -> ds_load_b128.
// Per K-step: prefetch tile k+1 async into alt buffer, compute tile k,
// s_wait_asynccnt 0, single workgroup barrier.
// =====================================================================
__device__ __forceinline__ void wmma_tile64x128(const __bf16* __restrict__ Ag, int lda,
                                                const __bf16* __restrict__ Bt, int ldbt, int K,
                                                __bf16* As, __bf16* Bs, v8f* c) {
    const int tid  = threadIdx.x;
    const int lane = tid & 31;
    const int wave = tid >> 5;
    const int sm   = wave & 3;
    const int scq  = (wave >> 2) << 2;       // 0 or 4 (subtile-column base)
    const int hi   = (lane >> 4) & 1;
    const int l15  = lane & 15;
    const int arow = tid >> 2, acol = (tid & 3) << 3;   // A: 64x32, 8 bf16/thread
    const int brow = tid >> 1, bcol = (tid & 1) << 4;   // B: 128x32, 16 bf16/thread

    const __bf16* agp = Ag + (size_t)arow * lda + acol;
    const __bf16* bgp = Bt + (size_t)brow * ldbt + bcol;
    const unsigned asl  = lds_off(&As[arow * PA + acol]);
    const unsigned bsl  = lds_off(&Bs[brow * PBK + bcol]);
    const unsigned abufB = ABUF * 2;         // bytes per A buffer
    const unsigned bbufB = BBUF * 2;         // bytes per B buffer

    const int aoff = (sm * 16 + l15) * PA + (hi << 3);      // K = 8*hi {+0..7, +16..23}
    int boff[4];
#pragma unroll
    for (int j = 0; j < 4; ++j)
        boff[j] = ((scq + j) * 16 + l15) * PBK + (hi << 3); // K = 8*hi {+0..7, +16..23}

#pragma unroll
    for (int j = 0; j < 4; ++j)
#pragma unroll
        for (int i = 0; i < 8; ++i) c[j][i] = 0.0f;

    // prologue: async-load tile 0 into buffer 0
    async_copy_b128(asl, agp);
    async_copy_b128(bsl, bgp);
    async_copy_b128(bsl + 16, bgp + 8);
    wait_asynccnt0();
    __syncthreads();

    for (int k0 = 0; k0 < K; k0 += 32) {
        const int cur = (k0 >> 5) & 1;
        const int nxt = cur ^ 1;

        // prefetch next K-tile into the alternate buffer (overlaps WMMA)
        if (k0 + 32 < K) {
            async_copy_b128(asl + nxt * abufB, agp + k0 + 32);
            async_copy_b128(bsl + nxt * bbufB, bgp + k0 + 32);
            async_copy_b128(bsl + nxt * bbufB + 16, bgp + k0 + 40);
        }

        // compute tile k0 from current buffer
        const __bf16* Ac = As + cur * ABUF;
        const __bf16* Bc = Bs + cur * BBUF;
        v8bf alo = *(const v8bf*)&Ac[aoff];
        v8bf ahi = *(const v8bf*)&Ac[aoff + 16];
        v16bf a = __builtin_shufflevector(alo, ahi, 0, 1, 2, 3, 4, 5, 6, 7,
                                          8, 9, 10, 11, 12, 13, 14, 15);
#pragma unroll
        for (int j = 0; j < 4; ++j) {
            v8bf blo = *(const v8bf*)&Bc[boff[j]];
            v8bf bhi = *(const v8bf*)&Bc[boff[j] + 16];
            v16bf b = __builtin_shufflevector(blo, bhi, 0, 1, 2, 3, 4, 5, 6, 7,
                                              8, 9, 10, 11, 12, 13, 14, 15);
            c[j] = __builtin_amdgcn_wmma_f32_16x16x32_bf16(false, a, false, b,
                                                           (short)0, c[j], false, false);
        }

        // own async writes done; barrier makes all waves' writes visible and
        // guarantees everyone finished reading 'cur' before it is overwritten
        wait_asynccnt0();
        __syncthreads();
    }
}

// C/D 16x16 f32 layout: VGPR r -> (M = r + 8*hi, N = lane&15)
#define TILE_EPILOGUE_VARS()                       \
    const int lane = threadIdx.x & 31;             \
    const int wave = threadIdx.x >> 5;             \
    const int sm   = wave & 3;                     \
    const int scq  = (wave >> 2) << 2;             \
    const int hi   = (lane >> 4) & 1;              \
    const int l15  = lane & 15;                    \
    const int rbase = sm * 16 + hi * 8;

// ---------------- shared GEMM1: hid = gelu(X @ W1 + b1), bf16 out ----------------
__global__ void __launch_bounds__(256) gemm_shared1(const __bf16* __restrict__ X,
                                                    const __bf16* __restrict__ W1t,
                                                    const float* __restrict__ b1,
                                                    __bf16* __restrict__ hid) {
    __shared__ __bf16 As[2 * ABUF];
    __shared__ __bf16 Bs[2 * BBUF];
    const __bf16* Ag = X + (size_t)blockIdx.x * 64 * D_MODEL;
    const __bf16* Bt = W1t + (size_t)blockIdx.y * 128 * D_MODEL;
    v8f c[4];
    wmma_tile64x128(Ag, D_MODEL, Bt, D_MODEL, D_MODEL, As, Bs, c);
    TILE_EPILOGUE_VARS();
#pragma unroll
    for (int j = 0; j < 4; ++j) {
        int cg = blockIdx.y * 128 + (scq + j) * 16 + l15;
        float bj = b1[cg];
#pragma unroll
        for (int r = 0; r < 8; ++r) {
            size_t row = (size_t)blockIdx.x * 64 + rbase + r;
            hid[row * HIDDEN + cg] = (__bf16)gelu_tanh(c[j][r] + bj);
        }
    }
}

// ---------------- shared GEMM2: out = hid @ W2 + b2, fp32 store ----------------
__global__ void __launch_bounds__(256) gemm_shared2(const __bf16* __restrict__ H,
                                                    const __bf16* __restrict__ W2t,
                                                    const float* __restrict__ b2,
                                                    float* __restrict__ out) {
    __shared__ __bf16 As[2 * ABUF];
    __shared__ __bf16 Bs[2 * BBUF];
    const __bf16* Ag = H + (size_t)blockIdx.x * 64 * HIDDEN;
    const __bf16* Bt = W2t + (size_t)blockIdx.y * 128 * HIDDEN;
    v8f c[4];
    wmma_tile64x128(Ag, HIDDEN, Bt, HIDDEN, HIDDEN, As, Bs, c);
    TILE_EPILOGUE_VARS();
#pragma unroll
    for (int j = 0; j < 4; ++j) {
        int cg = blockIdx.y * 128 + (scq + j) * 16 + l15;
        float bj = b2[cg];
#pragma unroll
        for (int r = 0; r < 8; ++r) {
            size_t row = (size_t)blockIdx.x * 64 + rbase + r;
            out[row * D_MODEL + cg] = c[j][r] + bj;
        }
    }
}

// ---------------- expert GEMM1: ehid = gelu(tok @ EW1[e] + eb1[e]) ----------------
__global__ void __launch_bounds__(256) gemm_exp1(const __bf16* __restrict__ Tok,
                                                 const __bf16* __restrict__ EW1t,
                                                 const float* __restrict__ eb1,
                                                 __bf16* __restrict__ ehid) {
    __shared__ __bf16 As[2 * ABUF];
    __shared__ __bf16 Bs[2 * BBUF];
    const int e = blockIdx.z;
    const __bf16* Ag = Tok + (size_t)e * CAP * D_MODEL + (size_t)blockIdx.x * 64 * D_MODEL;
    const __bf16* Bt = EW1t + (size_t)e * HIDDEN * D_MODEL + (size_t)blockIdx.y * 128 * D_MODEL;
    v8f c[4];
    wmma_tile64x128(Ag, D_MODEL, Bt, D_MODEL, D_MODEL, As, Bs, c);
    TILE_EPILOGUE_VARS();
    __bf16* hidE = ehid + (size_t)e * CAP * HIDDEN;
    const float* bE = eb1 + (size_t)e * HIDDEN;
#pragma unroll
    for (int j = 0; j < 4; ++j) {
        int cg = blockIdx.y * 128 + (scq + j) * 16 + l15;
        float bj = bE[cg];
#pragma unroll
        for (int r = 0; r < 8; ++r) {
            size_t row = (size_t)blockIdx.x * 64 + rbase + r;
            hidE[row * HIDDEN + cg] = (__bf16)gelu_tanh(c[j][r] + bj);
        }
    }
}

// ---------------- expert GEMM2: scatter-add score*(ehid @ EW2[e] + eb2[e]) -------
__global__ void __launch_bounds__(256) gemm_exp2(const __bf16* __restrict__ EHid,
                                                 const __bf16* __restrict__ EW2t,
                                                 const float* __restrict__ eb2,
                                                 const float* __restrict__ scores,
                                                 const int* __restrict__ tok_idx,
                                                 float* __restrict__ out) {
    __shared__ __bf16 As[2 * ABUF];
    __shared__ __bf16 Bs[2 * BBUF];
    const int e = blockIdx.z;
    const __bf16* Ag = EHid + (size_t)e * CAP * HIDDEN + (size_t)blockIdx.x * 64 * HIDDEN;
    const __bf16* Bt = EW2t + (size_t)e * D_MODEL * HIDDEN + (size_t)blockIdx.y * 128 * HIDDEN;
    v8f c[4];
    wmma_tile64x128(Ag, HIDDEN, Bt, HIDDEN, HIDDEN, As, Bs, c);
    TILE_EPILOGUE_VARS();
    const float* bE = eb2 + (size_t)e * D_MODEL;
#pragma unroll
    for (int j = 0; j < 4; ++j) {
        int cg = blockIdx.y * 128 + (scq + j) * 16 + l15;
        float bj = bE[cg];
#pragma unroll
        for (int r = 0; r < 8; ++r) {
            int crow = blockIdx.x * 64 + rbase + r;        // 0..CAP-1
            int tok = tok_idx[e * CAP + crow] & (NTOK - 1);
            float sc_w = scores[e * CAP + crow];
            atomicAdd(&out[(size_t)tok * D_MODEL + cg], (c[j][r] + bj) * sc_w);
        }
    }
}

// ---------------- host launcher ----------------
extern "C" void kernel_launch(void* const* d_in, const int* in_sizes, int n_in,
                              void* d_out, int out_size, void* d_ws, size_t ws_size,
                              hipStream_t stream) {
    const float* x      = (const float*)d_in[0];
    const float* gate_w = (const float*)d_in[1];
    const float* gate_b = (const float*)d_in[2];
    const float* temp   = (const float*)d_in[3];
    const float* sw1    = (const float*)d_in[4];
    const float* sb1    = (const float*)d_in[5];
    const float* sw2    = (const float*)d_in[6];
    const float* sb2    = (const float*)d_in[7];
    const float* ew1    = (const float*)d_in[8];
    const float* eb1    = (const float*)d_in[9];
    const float* ew2    = (const float*)d_in[10];
    const float* eb2    = (const float*)d_in[11];
    float* out = (float*)d_out;

    char* ws = (char*)d_ws;
    size_t off = 0;
    auto alloc = [&](size_t bytes) -> void* {
        void* p = ws + off;
        off = (off + bytes + 255) & ~(size_t)255;
        return p;
    };
    float*  probs   = (float*)alloc((size_t)NEXP * NTOK * 4);
    float*  scores  = (float*)alloc((size_t)NEXP * CAP * 4);
    int*    tok_idx = (int*)  alloc((size_t)NEXP * CAP * 4);
    int*    cnt     = (int*)  alloc(64);
    __bf16* xb      = (__bf16*)alloc((size_t)NTOK * D_MODEL * 2);
    __bf16* w1t     = (__bf16*)alloc((size_t)D_MODEL * HIDDEN * 2);   // [H][D]
    __bf16* w2t     = (__bf16*)alloc((size_t)HIDDEN * D_MODEL * 2);   // [D][H]
    __bf16* ew1t    = (__bf16*)alloc((size_t)NEXP * D_MODEL * HIDDEN * 2); // [E][H][D]
    __bf16* ew2t    = (__bf16*)alloc((size_t)NEXP * HIDDEN * D_MODEL * 2); // [E][D][H]
    __bf16* hidb    = (__bf16*)alloc((size_t)NTOK * HIDDEN * 2);
    __bf16* tokb    = (__bf16*)alloc((size_t)NEXP * CAP * D_MODEL * 2);
    __bf16* ehidb   = (__bf16*)alloc((size_t)NEXP * CAP * HIDDEN * 2);

    init_kernel<<<dim3(16), dim3(256), 0, stream>>>(cnt, tok_idx, scores);

    {   // x -> bf16 (row-major, used as A everywhere)
        long n = (long)NTOK * D_MODEL;
        cvt_bf16_kernel<<<dim3((unsigned)((n / 4 + 255) / 256)), dim3(256), 0, stream>>>(x, xb, n);
    }
    // weights -> bf16, TRANSPOSED to [N][K] so B tiles are K-contiguous
    cvt_transpose_kernel<<<dim3(D_MODEL / 32, HIDDEN / 32, 1), dim3(256), 0, stream>>>(sw1, w1t, D_MODEL, HIDDEN);
    cvt_transpose_kernel<<<dim3(HIDDEN / 32, D_MODEL / 32, 1), dim3(256), 0, stream>>>(sw2, w2t, HIDDEN, D_MODEL);
    cvt_transpose_kernel<<<dim3(D_MODEL / 32, HIDDEN / 32, NEXP), dim3(256), 0, stream>>>(ew1, ew1t, D_MODEL, HIDDEN);
    cvt_transpose_kernel<<<dim3(HIDDEN / 32, D_MODEL / 32, NEXP), dim3(256), 0, stream>>>(ew2, ew2t, HIDDEN, D_MODEL);

    router_kernel<<<dim3(NTOK / 8), dim3(256), 0, stream>>>(x, gate_w, gate_b, temp, probs);
    topk_kernel<<<dim3(NEXP), dim3(256), 0, stream>>>(probs, tok_idx, scores, cnt);
    gather_kernel<<<dim3(NEXP * CAP), dim3(256), 0, stream>>>(xb, tok_idx, tokb);

    // shared FFN
    gemm_shared1<<<dim3(NTOK / 64, HIDDEN / 128), dim3(256), 0, stream>>>(xb, w1t, sb1, hidb);
    gemm_shared2<<<dim3(NTOK / 64, D_MODEL / 128), dim3(256), 0, stream>>>(hidb, w2t, sb2, out);

    // routed experts
    gemm_exp1<<<dim3(CAP / 64, HIDDEN / 128, NEXP), dim3(256), 0, stream>>>(tokb, ew1t, eb1, ehidb);
    gemm_exp2<<<dim3(CAP / 64, D_MODEL / 128, NEXP), dim3(256), 0, stream>>>(ehidb, ew2t, eb2, scores, tok_idx, out);

    (void)in_sizes; (void)n_in; (void)out_size; (void)ws_size;
}
